// SwinBlock_3332894621925
// MI455X (gfx1250) — matrix operand order repaired and checked by reading, not compiled, not run on hardware
//
#include <hip/hip_runtime.h>
#include <hip/hip_bf16.h>
#include <math.h>

typedef __attribute__((ext_vector_type(16))) __bf16 v16bf;
typedef __attribute__((ext_vector_type(8)))  __bf16 v8bf;
typedef __attribute__((ext_vector_type(8)))  float  v8f;

#define HW   4096
#define CCH  128
#define BB   4

#define EPI_QKV       0
#define EPI_RESID_F32 1
#define EPI_GELU_BF16 2

// hardware bf16 convert (RTNE) — lowers to v_cvt_pk_bf16_f32
static __device__ __forceinline__ __bf16 f2bf(float f){
    return (__bf16)f;
}

// CDNA5 async DMA: global -> LDS, 16B per lane, tracked by ASYNCcnt (ISA 10, §15.18.3).
// LDS operand = low 32 bits of the generic pointer (flat->LDS uses addr[31:0], ISA §10.2).
static __device__ __forceinline__ void async_copy_b128(const void* gptr, void* lds){
    unsigned ldsOff = (unsigned)(size_t)lds;
    asm volatile("global_load_async_to_lds_b128 %0, %1, off"
                 :: "v"(ldsOff), "v"(gptr) : "memory");
}
static __device__ __forceinline__ void wait_async(){
    asm volatile("s_wait_asynccnt 0x0" ::: "memory");
}
// force distinct registers / issue order: all fragments materialized before first use
static __device__ __forceinline__ void pin4(v16bf& a, v16bf& b, v16bf& c, v16bf& d){
    asm volatile("" : "+v"(a), "+v"(b), "+v"(c), "+v"(d));
}
static __device__ __forceinline__ void pin3(v16bf& a, v16bf& b, v16bf& c){
    asm volatile("" : "+v"(a), "+v"(b), "+v"(c));
}

// A-matrix fragment (16x32 bf16): lanes 0-15 hold K {base..base+7, base+16..base+23}
// with base=0; lanes 16-31 with base=8 (per CDNA5 ISA 7.12.2).
static __device__ __forceinline__ v16bf frag_a(const __bf16* row, int lane){
    int base = (lane & 16) ? 8 : 0;
    v8bf lo = *(const v8bf*)(row + base);
    v8bf hi = *(const v8bf*)(row + base + 16);
    return __builtin_shufflevector(lo, hi, 0,1,2,3,4,5,6,7,8,9,10,11,12,13,14,15);
}
// B-matrix fragment (32x16 bf16): N = lane%16; lanes 0-15 K={0..7,16..23}, lanes 16-31 K={8..15,24..31}.
static __device__ __forceinline__ v16bf frag_b(const __bf16* row, int lane){
    int base = (lane & 16) ? 8 : 0;
    v8bf lo = *(const v8bf*)(row + base);
    v8bf hi = *(const v8bf*)(row + base + 16);
    return __builtin_shufflevector(lo, hi, 0,1,2,3,4,5,6,7,8,9,10,11,12,13,14,15);
}
static __device__ __forceinline__ v8f wmma_bf16(v16bf a, v16bf b, v8f c){
    return __builtin_amdgcn_wmma_f32_16x16x32_bf16(false, a, false, b, (short)0, c, false, false);
}

// ---------------- BN statistics: per-channel scale/shift ----------------
__global__ __launch_bounds__(256) void bn_stats_kernel(
    const float* __restrict__ x, const float* __restrict__ g, const float* __restrict__ bet,
    float* __restrict__ scale, float* __restrict__ shift)
{
    int c = blockIdx.x, t = threadIdx.x;
    float s = 0.f, ss = 0.f;
    for (int idx = t; idx < BB * HW; idx += 256){
        int b = idx >> 12, i = idx & (HW - 1);
        float v = x[((size_t)b * CCH + c) * HW + i];
        s += v; ss += v * v;
    }
    __shared__ float rs[256], rq[256];
    rs[t] = s; rq[t] = ss; __syncthreads();
    for (int off = 128; off; off >>= 1){
        if (t < off){ rs[t] += rs[t + off]; rq[t] += rq[t + off]; }
        __syncthreads();
    }
    if (t == 0){
        const float inv_n = 1.f / (float)(BB * HW);
        float mean = rs[0] * inv_n;
        float var  = rq[0] * inv_n - mean * mean;
        float sc = g[c] * rsqrtf(var + 1e-5f);
        scale[c] = sc;
        shift[c] = bet[c] - mean * sc;
    }
}

// ------------- fold BN into weights: W' = W*scale (bf16), b' = b + W@shift -------------
__global__ __launch_bounds__(128) void fold_kernel(
    const float* __restrict__ w, const float* __restrict__ bias,
    const float* __restrict__ scale, const float* __restrict__ shift,
    __bf16* __restrict__ wf, float* __restrict__ bf_, int K)
{
    int o = blockIdx.x, t = threadIdx.x;
    float part = 0.f;
    for (int k = t; k < K; k += 128){
        float wv = w[(size_t)o * K + k];
        wf[(size_t)o * K + k] = f2bf(wv * scale[k]);
        part += wv * shift[k];
    }
    __shared__ float r[128];
    r[t] = part; __syncthreads();
    for (int off = 64; off; off >>= 1){
        if (t < off) r[t] += r[t + off];
        __syncthreads();
    }
    if (t == 0) bf_[o] = bias[o] + r[0];
}

__global__ void cvt_kernel(const float* __restrict__ w, __bf16* __restrict__ o, int n){
    int i = blockIdx.x * 256 + threadIdx.x;
    if (i < n) o[i] = f2bf(w[i]);
}

// ---------------- generic bf16-WMMA GEMM: Y[b] = A[MxK] * X[b][KxHW] + bias ----------------
// Workgroup tile 64(M) x 128(N); each wave: one A-frag reused across 4 B-frags -> 4 WMMA/K-step.
__global__ __launch_bounds__(256) void gemm_kernel(
    const __bf16* __restrict__ A, const float* __restrict__ bias,
    const float* __restrict__ Bf, const __bf16* __restrict__ Bh,
    int M, int K,
    const float* __restrict__ resid, float* __restrict__ outF, __bf16* __restrict__ outH,
    int mode, float qscale)
{
    __shared__ __align__(16) __bf16 As[64][32];    // [m][k]
    __shared__ __align__(16) __bf16 Bs[128][32];   // [n][k]  (transposed stage)
    int b = blockIdx.z;
    int mBase = blockIdx.x * 64;
    int nBase = blockIdx.y * 128;
    int t = threadIdx.x, lane = t & 31, w = t >> 5;
    int lh = lane & 15;
    size_t bOffIn = (size_t)b * K * HW;

    v8f acc[4] = {};
    int mT = (w >> 1) * 16;     // wave's local m-tile row
    int nG = (w & 1) * 64;      // wave covers four 16-wide n-tiles

    for (int kt = 0; kt < K; kt += 32){
        __syncthreads();
        {   // stage A tile (64x32) via async DMA (pure copy, no VGPR round-trip)
            int r = t >> 2, cs = (t & 3) * 8;
            async_copy_b128(A + (size_t)(mBase + r) * K + kt + cs, &As[r][cs]);
            // stage B tile transposed (n-major), converting f32 -> bf16 on the fly
            int k = t & 31, ns = (t >> 5) * 16;
            if (Bf){
                const float* p = Bf + bOffIn + (size_t)(kt + k) * HW + nBase + ns;
                #pragma unroll
                for (int q4 = 0; q4 < 4; q4++){
                    float4 u = *(const float4*)(p + q4 * 4);
                    Bs[ns + q4*4 + 0][k] = f2bf(u.x);
                    Bs[ns + q4*4 + 1][k] = f2bf(u.y);
                    Bs[ns + q4*4 + 2][k] = f2bf(u.z);
                    Bs[ns + q4*4 + 3][k] = f2bf(u.w);
                }
                if (kt + 32 < K) __builtin_prefetch(p + 32 * HW, 0, 3);
            } else {
                const __bf16* p = Bh + bOffIn + (size_t)(kt + k) * HW + nBase + ns;
                v8bf u0 = *(const v8bf*)p, u1 = *(const v8bf*)(p + 8);
                #pragma unroll
                for (int j = 0; j < 8; j++){ Bs[ns + j][k] = u0[j]; Bs[ns + 8 + j][k] = u1[j]; }
                if (kt + 32 < K) __builtin_prefetch(p + 32 * HW, 0, 3);
            }
        }
        wait_async();
        __syncthreads();
        // all fragment loads issue as one clause (single dscnt wait), WMMAs back-to-back
        v16bf a  = frag_a(&As[mT + lh][0], lane);
        v16bf b0 = frag_b(&Bs[nG +  0 + lh][0], lane);
        v16bf b1 = frag_b(&Bs[nG + 16 + lh][0], lane);
        v16bf b2 = frag_b(&Bs[nG + 32 + lh][0], lane);
        v16bf b3 = frag_b(&Bs[nG + 48 + lh][0], lane);
        pin4(b0, b1, b2, b3);
        acc[0] = wmma_bf16(a, b0, acc[0]);
        acc[1] = wmma_bf16(a, b1, acc[1]);
        acc[2] = wmma_bf16(a, b2, acc[2]);
        acc[3] = wmma_bf16(a, b3, acc[3]);
    }

    // epilogue (C layout: VGPR r -> M = r (+8 for lanes 16..31), N = lane%16)
    int hi = (lane & 16) ? 8 : 0;
    #pragma unroll
    for (int r = 0; r < 8; r++){
        int m = mBase + mT + r + hi;
        float bv = bias[m];
        size_t rowOff = ((size_t)b * M + m) * HW;
        #pragma unroll
        for (int i = 0; i < 4; i++){
            float y = acc[i][r] + bv;
            size_t idx = rowOff + nBase + nG + i * 16 + lh;
            if (mode == EPI_QKV){
                float s = (m < 128) ? qscale : 1.f;   // fold attention scale into q
                outH[idx] = f2bf(y * s);
            } else if (mode == EPI_RESID_F32){
                outF[idx] = y + resid[idx];
            } else { // exact GELU
                outH[idx] = f2bf(0.5f * y * (1.f + erff(y * 0.70710678f)));
            }
        }
    }
}

// ---------------- flash attention: per (query-block=128, head, batch) ----------------
// 128-key chunks: 16 WMMA per barrier pair, 56 KB LDS of the 320 KB WGP pool.
__global__ __launch_bounds__(256) void attn_kernel(
    const __bf16* __restrict__ qkv, __bf16* __restrict__ out)
{
    __shared__ __align__(16) __bf16 qT[128][32];      // [i][d]
    __shared__ __align__(16) __bf16 kT[128][32];      // [j][d]
    __shared__ __align__(16) __bf16 vS[32][128];      // [d][j]
    __shared__ __align__(16) __bf16 pS[8][16][128];   // per-wave P tile [i][j]
    int iBlk = blockIdx.x * 128;
    int head = blockIdx.y;
    int b    = blockIdx.z;
    int t = threadIdx.x, lane = t & 31, w = t >> 5;
    int lh = lane & 15;
    int hi = (lane & 16) ? 8 : 0;

    const __bf16* qBase = qkv + ((size_t)b * 384 + head * 32) * HW;
    const __bf16* kBase = qBase + (size_t)128 * HW;
    const __bf16* vBase = qBase + (size_t)256 * HW;

    {   // stage Q^T for the whole block (q already scaled by 1/sqrt(hd))
        int d = t & 31, is = (t >> 5) * 16;
        const __bf16* p = qBase + (size_t)d * HW + iBlk + is;
        v8bf u = *(const v8bf*)p, v = *(const v8bf*)(p + 8);
        #pragma unroll
        for (int j = 0; j < 8; j++){ qT[is + j][d] = u[j]; qT[is + 8 + j][d] = v[j]; }
    }
    __syncthreads();
    v16bf aQ = frag_a(&qT[w * 16 + lh][0], lane);

    v8f oA = {}, oB = {};
    float mrow[8], lrow[8];
    #pragma unroll
    for (int r = 0; r < 8; r++){ mrow[r] = -1e30f; lrow[r] = 0.f; }

    for (int j0 = 0; j0 < HW; j0 += 128){
        __syncthreads();
        {   // stage K (transposed, VGPR path) + V (async DMA direct copy)
            int d = t >> 3, js = (t & 7) * 16;
            const __bf16* kp = kBase + (size_t)d * HW + j0 + js;
            v8bf k0 = *(const v8bf*)kp, k1 = *(const v8bf*)(kp + 8);
            #pragma unroll
            for (int j = 0; j < 8; j++){ kT[js + j][d] = k0[j]; kT[js + 8 + j][d] = k1[j]; }
            const __bf16* vp = vBase + (size_t)d * HW + j0 + js;
            async_copy_b128(vp,     &vS[d][js]);
            async_copy_b128(vp + 8, &vS[d][js + 8]);
            if (j0 + 128 < HW) __builtin_prefetch(kp + 128, 0, 3);
        }
        wait_async();
        __syncthreads();

        // S = Q^T K  (8 j-tiles of 16, K=d=32 per WMMA; fragments pinned in groups of 4)
        v8f s[8];
        #pragma unroll
        for (int g = 0; g < 2; g++){
            v16bf k0 = frag_b(&kT[(g*4 + 0) * 16 + lh][0], lane);
            v16bf k1 = frag_b(&kT[(g*4 + 1) * 16 + lh][0], lane);
            v16bf k2 = frag_b(&kT[(g*4 + 2) * 16 + lh][0], lane);
            v16bf k3 = frag_b(&kT[(g*4 + 3) * 16 + lh][0], lane);
            pin4(k0, k1, k2, k3);
            v8f z0 = {}, z1 = {}, z2 = {}, z3 = {};
            s[g*4 + 0] = wmma_bf16(aQ, k0, z0);
            s[g*4 + 1] = wmma_bf16(aQ, k1, z1);
            s[g*4 + 2] = wmma_bf16(aQ, k2, z2);
            s[g*4 + 3] = wmma_bf16(aQ, k3, z3);
        }

        // online softmax (row = r + hi within wave tile; reduce across 16 lanes/half)
        #pragma unroll
        for (int r = 0; r < 8; r++){
            float mx = s[0][r];
            #pragma unroll
            for (int tt = 1; tt < 8; tt++) mx = fmaxf(mx, s[tt][r]);
            #pragma unroll
            for (int msk = 1; msk < 16; msk <<= 1) mx = fmaxf(mx, __shfl_xor(mx, msk, 32));
            float mn  = fmaxf(mrow[r], mx);
            float fac = __expf(mrow[r] - mn);
            float pv[8], rs = 0.f;
            #pragma unroll
            for (int tt = 0; tt < 8; tt++){ pv[tt] = __expf(s[tt][r] - mn); rs += pv[tt]; }
            #pragma unroll
            for (int msk = 1; msk < 16; msk <<= 1) rs += __shfl_xor(rs, msk, 32);
            lrow[r] = lrow[r] * fac + rs;
            mrow[r] = mn;
            oA[r] *= fac; oB[r] *= fac;
            int mi = r + hi;                 // C-layout row -> LDS row
            #pragma unroll
            for (int tt = 0; tt < 8; tt++) pS[w][mi][tt * 16 + lh] = f2bf(pv[tt]);
        }
        // O += P V   (P re-fragmented via per-wave LDS; same-wave DS ops are in-order)
        #pragma unroll
        for (int kc = 0; kc < 4; kc++){
            v16bf pA = frag_a(&pS[w][lh][kc * 32], lane);
            v16bf v0 = frag_b(&vS[ 0 + lh][kc * 32], lane);
            v16bf v1 = frag_b(&vS[16 + lh][kc * 32], lane);
            pin3(pA, v0, v1);
            oA = wmma_bf16(pA, v0, oA);
            oB = wmma_bf16(pA, v1, oB);
        }
    }

    // finalize: O /= l ; store out[b][head*32+d][i] as bf16
    size_t outBase = ((size_t)b * 128 + head * 32) * HW;
    #pragma unroll
    for (int r = 0; r < 8; r++){
        float inv = 1.f / lrow[r];
        int iG = iBlk + w * 16 + r + hi;
        out[outBase + (size_t)( 0 + lh) * HW + iG] = f2bf(oA[r] * inv);
        out[outBase + (size_t)(16 + lh) * HW + iG] = f2bf(oB[r] * inv);
    }
}

extern "C" void kernel_launch(void* const* d_in, const int* in_sizes, int n_in,
                              void* d_out, int out_size, void* d_ws, size_t ws_size,
                              hipStream_t stream)
{
    (void)in_sizes; (void)n_in; (void)out_size; (void)ws_size;
    const float* x      = (const float*)d_in[0];
    const float* bn1_g  = (const float*)d_in[1];
    const float* bn1_b  = (const float*)d_in[2];
    const float* qkv_w  = (const float*)d_in[3];
    const float* qkv_b  = (const float*)d_in[4];
    const float* proj_w = (const float*)d_in[5];
    const float* proj_b = (const float*)d_in[6];
    const float* bn2_g  = (const float*)d_in[7];
    const float* bn2_b  = (const float*)d_in[8];
    const float* mlp_w1 = (const float*)d_in[9];
    const float* mlp_b1 = (const float*)d_in[10];
    const float* mlp_w2 = (const float*)d_in[11];
    const float* mlp_b2 = (const float*)d_in[12];
    float* out = (float*)d_out;

    char* ws = (char*)d_ws;
    size_t off = 0;
    auto alloc = [&](size_t bytes) -> char* {
        off = (off + 255) & ~(size_t)255;
        char* p = ws + off;
        off += bytes;
        return p;
    };
    float*  scale1  = (float*)alloc(CCH * 4);
    float*  shift1  = (float*)alloc(CCH * 4);
    float*  scale2  = (float*)alloc(CCH * 4);
    float*  shift2  = (float*)alloc(CCH * 4);
    __bf16* qkvw    = (__bf16*)alloc((size_t)384 * 128 * 2);
    float*  qkvb    = (float*)alloc(384 * 4);
    __bf16* projw   = (__bf16*)alloc((size_t)128 * 128 * 2);
    __bf16* w1f     = (__bf16*)alloc((size_t)512 * 128 * 2);
    float*  b1f     = (float*)alloc(512 * 4);
    __bf16* w2f     = (__bf16*)alloc((size_t)128 * 512 * 2);
    __bf16* qkvAct  = (__bf16*)alloc((size_t)BB * 384 * HW * 2);
    __bf16* attnOut = (__bf16*)alloc((size_t)BB * 128 * HW * 2);
    float*  xres    = (float*)alloc((size_t)BB * 128 * HW * 4);
    __bf16* m1      = (__bf16*)alloc((size_t)BB * 512 * HW * 2);

    const float qscale = 0.17677669529663687f;  // 1/sqrt(32)

    // BN1 stats + fold into qkv weights
    bn_stats_kernel<<<128, 256, 0, stream>>>(x, bn1_g, bn1_b, scale1, shift1);
    fold_kernel<<<384, 128, 0, stream>>>(qkv_w, qkv_b, scale1, shift1, qkvw, qkvb, 128);
    // qkv = W' x + b'   (q rows pre-scaled)
    gemm_kernel<<<dim3(6, 32, BB), 256, 0, stream>>>(qkvw, qkvb, x, nullptr, 384, 128,
                                                     nullptr, nullptr, qkvAct, EPI_QKV, qscale);
    // flash attention
    attn_kernel<<<dim3(HW / 128, 4, BB), 256, 0, stream>>>(qkvAct, attnOut);
    // proj + residual -> xres (fp32)
    cvt_kernel<<<(128 * 128) / 256, 256, 0, stream>>>(proj_w, projw, 128 * 128);
    gemm_kernel<<<dim3(2, 32, BB), 256, 0, stream>>>(projw, proj_b, nullptr, attnOut, 128, 128,
                                                     x, xres, nullptr, EPI_RESID_F32, 1.f);
    // BN2 stats + fold into mlp_w1; MLP
    bn_stats_kernel<<<128, 256, 0, stream>>>(xres, bn2_g, bn2_b, scale2, shift2);
    fold_kernel<<<512, 128, 0, stream>>>(mlp_w1, mlp_b1, scale2, shift2, w1f, b1f, 128);
    gemm_kernel<<<dim3(8, 32, BB), 256, 0, stream>>>(w1f, b1f, xres, nullptr, 512, 128,
                                                     nullptr, nullptr, m1, EPI_GELU_BF16, 1.f);
    cvt_kernel<<<(128 * 512) / 256, 256, 0, stream>>>(mlp_w2, w2f, 128 * 512);
    gemm_kernel<<<dim3(2, 32, BB), 256, 0, stream>>>(w2f, mlp_b2, nullptr, m1, 128, 512,
                                                     xres, out, nullptr, EPI_RESID_F32, 1.f);
}
